// MultiHeadedSelfAttention_59631325937798
// MI455X (gfx1250) — matrix operand order repaired and checked by reading, not compiled, not run on hardware
//
#include <hip/hip_runtime.h>

// ---------------------------------------------------------------------------
// MI455X (gfx1250) multi-head self-attention: bf16 WMMA + f32 accumulation.
// B=4, S=2048, D=1024, H=16, head_dim=64.
// V stored transposed (B,H,64,S).  Attention blocks cooperatively stage K/V
// tiles into LDS (async global->LDS when available), double-buffered.
// ---------------------------------------------------------------------------

typedef __attribute__((ext_vector_type(16))) __bf16 v16bf;
typedef __attribute__((ext_vector_type(8)))  float v8f;
typedef __attribute__((ext_vector_type(4)))  int   v4i;

union BFrag { v16bf v; unsigned int u[8]; unsigned short s[16]; };
union FAcc  { v8f v; float f[8]; };
union Pack8 { uint4 q; unsigned short s[8]; };
union Pack4 { uint2 q; unsigned short s[4]; };

__device__ __forceinline__ unsigned short f2bf(float x) {
    unsigned int u = __builtin_bit_cast(unsigned int, x);
    u = (u + 0x7FFFu + ((u >> 16) & 1u)) >> 16;   // round-to-nearest-even
    return (unsigned short)u;
}

// ---- async global->LDS (CDNA5) with synchronous fallback -------------------
#if __has_builtin(__builtin_amdgcn_global_load_async_to_lds_b128)
#define USE_ASYNC 1
#else
#define USE_ASYNC 0
#endif

typedef __attribute__((address_space(1))) v4i as1_v4i;
typedef __attribute__((address_space(3))) v4i as3_v4i;

__device__ __forceinline__ void copy16(const unsigned short* g, unsigned short* l) {
#if USE_ASYNC
    __builtin_amdgcn_global_load_async_to_lds_b128(
        (as1_v4i*)g, (as3_v4i*)l, 0, 0);
#else
    *(uint4*)l = *(const uint4*)g;
#endif
}

__device__ __forceinline__ void async_wait0() {
#if USE_ASYNC
#if __has_builtin(__builtin_amdgcn_s_wait_asynccnt)
    __builtin_amdgcn_s_wait_asynccnt(0);
#else
    asm volatile("s_wait_asynccnt 0x0" ::: "memory");
#endif
#endif
}

// ---------------------------------------------------------------------------
// Kernel 1: W (D,D) f32 -> Wt bf16 with Wt[k*D + n] = W[n*D + k]  (y = x @ W^T)
// ---------------------------------------------------------------------------
__global__ __launch_bounds__(256) void convert_weights(
    const float* __restrict__ Wq, const float* __restrict__ Wk,
    const float* __restrict__ Wv,
    unsigned short* __restrict__ WtQ, unsigned short* __restrict__ WtK,
    unsigned short* __restrict__ WtV) {
    const float* W = (blockIdx.y == 0) ? Wq : (blockIdx.y == 1 ? Wk : Wv);
    unsigned short* Wt = (blockIdx.y == 0) ? WtQ : (blockIdx.y == 1 ? WtK : WtV);
    int idx = blockIdx.x * 256 + threadIdx.x;     // 0 .. 1024*1024-1
    int k = idx >> 10, n = idx & 1023;
    Wt[idx] = f2bf(W[n * 1024 + k]);
}

// ---------------------------------------------------------------------------
// Kernel 2: QKV projection GEMM.  y[m,e] = sum_d x[m,d]*Wt[d,e] + bias[e]
// Block = 256 thr (8 waves), tile M=128 x N=64, K-step 32.
// Q (scaled by 1/8), K stored (B,H,S,64); V stored transposed (B,H,64,S).
// ---------------------------------------------------------------------------
__global__ __launch_bounds__(256) void qkv_gemm(
    const float* __restrict__ x,
    const unsigned short* __restrict__ WtQ, const unsigned short* __restrict__ WtK,
    const unsigned short* __restrict__ WtV,
    const float* __restrict__ bq, const float* __restrict__ bk,
    const float* __restrict__ bv,
    unsigned short* __restrict__ Qo, unsigned short* __restrict__ Ko,
    unsigned short* __restrict__ Vo) {
    __shared__ unsigned short Alds[128 * 36];   // [row][k] bf16, pad 36 (8B-aligned rows)
    __shared__ unsigned short Blds[64 * 36];    // [n][k]  bf16 (transposed)

    const int z = blockIdx.z;
    const unsigned short* Wt = (z == 0) ? WtQ : (z == 1 ? WtK : WtV);
    const float* bias        = (z == 0) ? bq  : (z == 1 ? bk  : bv);
    unsigned short* Out      = (z == 0) ? Qo  : (z == 1 ? Ko  : Vo);
    const float oscale       = (z == 0) ? 0.125f : 1.0f;   // 1/sqrt(head_dim)

    const int m0 = blockIdx.x * 128;
    const int n0 = blockIdx.y * 64;
    const int tid = threadIdx.x;
    const int wave = tid >> 5, lane = tid & 31, l16 = lane & 15, half = lane >> 4;

    FAcc acc[4];
#pragma unroll
    for (int i = 0; i < 4; ++i)
#pragma unroll
        for (int r = 0; r < 8; ++r) acc[i].f[r] = 0.f;

    for (int kk = 0; kk < 1024; kk += 32) {
        // stage A: x tile 128x32 f32 -> bf16 LDS; float4 loads, 8B LDS stores
#pragma unroll
        for (int i = 0; i < 4; ++i) {
            int c = tid + i * 256;            // 0..1023 float4 chunks
            int r = c >> 3, col = (c & 7) * 4;
            float4 xv = *(const float4*)&x[(m0 + r) * 1024 + kk + col];
            Pack4 p;
            p.s[0] = f2bf(xv.x); p.s[1] = f2bf(xv.y);
            p.s[2] = f2bf(xv.z); p.s[3] = f2bf(xv.w);
            *(uint2*)&Alds[r * 36 + col] = p.q;
        }
        // stage B transposed: Blds[n][k] = Wt[kk+k][n0+n]; uint reads (2 n's)
#pragma unroll
        for (int i = 0; i < 4; ++i) {
            int c = tid + i * 256;            // 0..1023 uint chunks
            int kl = c >> 5, n2 = (c & 31) * 2;
            unsigned int w = *(const unsigned int*)&Wt[(kk + kl) * 1024 + n0 + n2];
            Blds[n2 * 36 + kl]       = (unsigned short)(w & 0xFFFFu);
            Blds[(n2 + 1) * 36 + kl] = (unsigned short)(w >> 16);
        }
        __syncthreads();

        // A fragment: lane m=l16, VGPR j -> K = 2j + 16*half
        BFrag a;
#pragma unroll
        for (int j = 0; j < 8; ++j) {
            int kj = 2 * j + 16 * half;
            a.u[j] = *(const unsigned int*)&Alds[(wave * 16 + l16) * 36 + kj];
        }
#pragma unroll
        for (int nt = 0; nt < 4; ++nt) {
            // B fragment: lane n=l16, VGPR j -> K = 2j + 16*half
            BFrag b;
#pragma unroll
            for (int j = 0; j < 8; ++j)
                b.u[j] = *(const unsigned int*)&Blds[(nt * 16 + l16) * 36 + 2 * j + 16 * half];
            acc[nt].v = __builtin_amdgcn_wmma_f32_16x16x32_bf16(
                false, a.v, false, b.v, (short)0, acc[nt].v, false, false);
        }
        __syncthreads();
    }

    // epilogue: bias + scale; Q/K normal layout, V transposed with 128-bit stores
    const int bb = m0 >> 11;                  // batch constant per block (2048%128==0)
#pragma unroll
    for (int nt = 0; nt < 4; ++nt) {
        int e = n0 + nt * 16 + l16;
        float bias_e = bias[e];
        int h = e >> 6, dd = e & 63;
        if (z == 2) {
            // V transposed: Vt[((bb*16+h)*64+dd)*2048 + s]; 8 consecutive s per lane
            Pack8 p;
#pragma unroll
            for (int r = 0; r < 8; ++r) p.s[r] = f2bf(acc[nt].f[r] + bias_e);
            int s0 = (m0 & 2047) + wave * 16 + 8 * half;   // r=0 row
            *(uint4*)&Out[(((size_t)(bb * 16 + h) * 64 + dd) * 2048) + s0] = p.q;
        } else {
#pragma unroll
            for (int r = 0; r < 8; ++r) {
                int m = m0 + wave * 16 + r + 8 * half;     // C layout: VGPR r -> row r / r+8
                int s = m & 2047;
                float y = (acc[nt].f[r] + bias_e) * oscale;
                Out[(((size_t)(bb * 16 + h) * 2048) + s) * 64 + dd] = f2bf(y);
            }
        }
    }
}

// ---------------------------------------------------------------------------
// Kernel 3: flash attention.  Block = 4 waves = 128 queries of ONE (b,h).
// K/V 32-key tiles staged cooperatively into LDS (async), double-buffered;
// each wave computes 32 queries; K/V fragments reused by 2 WMMAs each.
// ---------------------------------------------------------------------------
__global__ __launch_bounds__(128) void attention(
    const unsigned short* __restrict__ Q, const unsigned short* __restrict__ K,
    const unsigned short* __restrict__ Vt, float* __restrict__ out) {
    __shared__ unsigned short Ktile[2][32 * 64];  // [key][d]   4KB per buffer
    __shared__ unsigned short Vtile[2][64 * 32];  // [d][key]   4KB per buffer
    __shared__ unsigned short Plds[4][32 * 34];   // per-wave P [row][key]

    const int tid = threadIdx.x;
    const int wave = tid >> 5, lane = tid & 31, l16 = lane & 15, half = lane >> 4;
    const int bh = blockIdx.x >> 4;               // 16 blocks of 128 queries per (b,h)
    const int q0 = (blockIdx.x & 15) * 128 + wave * 32;
    const int bb = bh >> 4, h = bh & 15;

    const unsigned short* Qb  = Q  + (size_t)bh * 2048 * 64;
    const unsigned short* Kb  = K  + (size_t)bh * 2048 * 64;
    const unsigned short* Vtb = Vt + (size_t)bh * 64 * 2048;

    // cooperative stage of one 32-key step into buffer `buf`
    auto stage = [&](int buf, int kt) {
        const unsigned short* kg = Kb + (size_t)kt * 64;      // 4KB contiguous
#pragma unroll
        for (int i = 0; i < 2; ++i) {
            int c = tid + i * 128;                            // 0..255 16B chunks
            copy16(kg + c * 8, &Ktile[buf][c * 8]);
        }
#pragma unroll
        for (int i = 0; i < 2; ++i) {
            int c = tid + i * 128;                            // 0..255 16B chunks
            int d = c >> 2, part = (c & 3) * 8;               // 4 chunks per d-row
            copy16(Vtb + (size_t)d * 2048 + kt + part, &Vtile[buf][d * 32 + part]);
        }
    };

    // Q fragments (A layout): [q-subtile][d-half 0..31 / 32..63]
    BFrag qa[2][2];
#pragma unroll
    for (int qt = 0; qt < 2; ++qt) {
        const unsigned short* qrow = Qb + (q0 + qt * 16 + l16) * 64;
#pragma unroll
        for (int j = 0; j < 8; ++j) {
            int kj = 2 * j + 16 * half;
            qa[qt][0].u[j] = *(const unsigned int*)&qrow[kj];
            qa[qt][1].u[j] = *(const unsigned int*)&qrow[kj + 32];
        }
    }

    FAcc acc[2][4];
    float rmax[2][8], rsum[2][8];
#pragma unroll
    for (int qt = 0; qt < 2; ++qt) {
#pragma unroll
        for (int i = 0; i < 4; ++i)
#pragma unroll
            for (int r = 0; r < 8; ++r) acc[qt][i].f[r] = 0.f;
#pragma unroll
        for (int r = 0; r < 8; ++r) { rmax[qt][r] = -1e30f; rsum[qt][r] = 0.f; }
    }

    unsigned short* P = Plds[wave];

    stage(0, 0);
    async_wait0();
    __syncthreads();

    int buf = 0;
    for (int kt = 0; kt < 2048; kt += 32) {
        if (kt + 32 < 2048) stage(buf ^ 1, kt + 32);   // overlap copy with compute

        // ---- scores: 2 key n-tiles x 2 q-subtiles; K frag reused across qt ----
        FAcc sc[2][2], zc;   // sc[qt][nt]
#pragma unroll
        for (int r = 0; r < 8; ++r) zc.f[r] = 0.f;
#pragma unroll
        for (int nt = 0; nt < 2; ++nt) {
            const unsigned short* krow = &Ktile[buf][(nt * 16 + l16) * 64];
            BFrag b;
#pragma unroll
            for (int j = 0; j < 8; ++j)
                b.u[j] = *(const unsigned int*)&krow[2 * j + 16 * half];
            sc[0][nt].v = __builtin_amdgcn_wmma_f32_16x16x32_bf16(false, qa[0][0].v, false, b.v, (short)0, zc.v, false, false);
            sc[1][nt].v = __builtin_amdgcn_wmma_f32_16x16x32_bf16(false, qa[1][0].v, false, b.v, (short)0, zc.v, false, false);
#pragma unroll
            for (int j = 0; j < 8; ++j)
                b.u[j] = *(const unsigned int*)&krow[2 * j + 16 * half + 32];
            sc[0][nt].v = __builtin_amdgcn_wmma_f32_16x16x32_bf16(false, qa[0][1].v, false, b.v, (short)0, sc[0][nt].v, false, false);
            sc[1][nt].v = __builtin_amdgcn_wmma_f32_16x16x32_bf16(false, qa[1][1].v, false, b.v, (short)0, sc[1][nt].v, false, false);
        }

        // ---- online softmax per q-subtile ----
#pragma unroll
        for (int qt = 0; qt < 2; ++qt) {
#pragma unroll
            for (int r = 0; r < 8; ++r) {
                float a = sc[qt][0].f[r], c = sc[qt][1].f[r];
                float mx = fmaxf(a, c);
                mx = fmaxf(mx, __shfl_xor(mx, 1));
                mx = fmaxf(mx, __shfl_xor(mx, 2));
                mx = fmaxf(mx, __shfl_xor(mx, 4));
                mx = fmaxf(mx, __shfl_xor(mx, 8));
                float nm = fmaxf(rmax[qt][r], mx);
                float corr = __expf(rmax[qt][r] - nm);
                rmax[qt][r] = nm;
                float p0 = __expf(a - nm), p1 = __expf(c - nm);
                float ls = p0 + p1;
                ls += __shfl_xor(ls, 1);
                ls += __shfl_xor(ls, 2);
                ls += __shfl_xor(ls, 4);
                ls += __shfl_xor(ls, 8);
                rsum[qt][r] = rsum[qt][r] * corr + ls;
#pragma unroll
                for (int dt = 0; dt < 4; ++dt) acc[qt][dt].f[r] *= corr;
                // C layout -> LDS row-major for A-layout reload
                P[(qt * 16 + r + 8 * half) * 34 + l16]      = f2bf(p0);
                P[(qt * 16 + r + 8 * half) * 34 + 16 + l16] = f2bf(p1);
            }
        }
        asm volatile("s_wait_dscnt 0" ::: "memory");   // intra-wave LDS store->load

        // ---- reload P as A fragments ----
        BFrag pa[2];
#pragma unroll
        for (int qt = 0; qt < 2; ++qt)
#pragma unroll
            for (int j = 0; j < 8; ++j) {
                int kj = 2 * j + 16 * half;
                pa[qt].u[j] = *(const unsigned int*)&P[(qt * 16 + l16) * 34 + kj];
            }

        // ---- O += P @ V : 4 d-tiles; V frag (contiguous keys) reused across qt ----
#pragma unroll
        for (int dt = 0; dt < 4; ++dt) {
            const unsigned short* vrow = &Vtile[buf][(dt * 16 + l16) * 32];
            BFrag b;
#pragma unroll
            for (int j = 0; j < 8; ++j)
                b.u[j] = *(const unsigned int*)&vrow[2 * j + 16 * half];
            acc[0][dt].v = __builtin_amdgcn_wmma_f32_16x16x32_bf16(
                false, pa[0].v, false, b.v, (short)0, acc[0][dt].v, false, false);
            acc[1][dt].v = __builtin_amdgcn_wmma_f32_16x16x32_bf16(
                false, pa[1].v, false, b.v, (short)0, acc[1][dt].v, false, false);
        }

        async_wait0();       // this wave's next-tile copies landed
        __syncthreads();     // all waves done reading buf + staging buf^1
        buf ^= 1;
    }

    // ---- finalize: divide by row sums, store f32 (B,S,D) merged heads ----
#pragma unroll
    for (int qt = 0; qt < 2; ++qt)
#pragma unroll
        for (int r = 0; r < 8; ++r) {
            float inv = 1.f / rsum[qt][r];
            int s = q0 + qt * 16 + r + 8 * half;
            float* orow = out + ((size_t)(bb * 2048 + s)) * 1024 + h * 64;
#pragma unroll
            for (int dt = 0; dt < 4; ++dt)
                orow[dt * 16 + l16] = acc[qt][dt].f[r] * inv;
        }
}

// ---------------------------------------------------------------------------
// Launch
// ---------------------------------------------------------------------------
extern "C" void kernel_launch(void* const* d_in, const int* in_sizes, int n_in,
                              void* d_out, int out_size, void* d_ws, size_t ws_size,
                              hipStream_t stream) {
    const float* x  = (const float*)d_in[0];
    const float* Wq = (const float*)d_in[1];
    const float* bq = (const float*)d_in[2];
    const float* Wk = (const float*)d_in[3];
    const float* bk = (const float*)d_in[4];
    const float* Wv = (const float*)d_in[5];
    const float* bv = (const float*)d_in[6];

    unsigned short* ws  = (unsigned short*)d_ws;
    const size_t WSZ = (size_t)1024 * 1024;          // weight matrix elems
    const size_t QSZ = (size_t)4 * 16 * 2048 * 64;   // 8,388,608 elems
    unsigned short* WtQ = ws;
    unsigned short* WtK = WtQ + WSZ;
    unsigned short* WtV = WtK + WSZ;
    unsigned short* Qb  = WtV + WSZ;
    unsigned short* Kb  = Qb + QSZ;
    unsigned short* Vtb = Kb + QSZ;                  // total ~54 MB of d_ws

    convert_weights<<<dim3(4096, 3), 256, 0, stream>>>(Wq, Wk, Wv, WtQ, WtK, WtV);
    qkv_gemm<<<dim3(64, 16, 3), 256, 0, stream>>>(x, WtQ, WtK, WtV, bq, bk, bv,
                                                  Qb, Kb, Vtb);
    attention<<<dim3(1024), 128, 0, stream>>>(Qb, Kb, Vtb, (float*)d_out);
}
